// MultiHeadAttention_31155692765433
// MI455X (gfx1250) — matrix operand order repaired and checked by reading, not compiled, not run on hardware
//
#include <hip/hip_runtime.h>
#include <hip/hip_bf16.h>

typedef __attribute__((ext_vector_type(16))) __bf16 v16bf;
typedef __attribute__((ext_vector_type(8)))  __bf16 v8bf;
typedef __attribute__((ext_vector_type(8)))  float  v8f;
typedef __attribute__((ext_vector_type(4)))  unsigned int u32x4;
typedef __attribute__((ext_vector_type(8)))  int i32x8;
typedef __attribute__((ext_vector_type(4)))  int i32x4;

#define BATCH 4
#define SEQ   1024
#define DIM   1024
#define NH    16
#define HD    64
#define NROW  (BATCH * SEQ)   // 4096

#if __has_builtin(__builtin_amdgcn_tensor_load_to_lds)
#define USE_TDM 1
#else
#define USE_TDM 0
#endif

// Combine two contiguous 8x bf16 (16B) chunks into one 16-element WMMA fragment.
static __device__ __forceinline__ v16bf ld2(const __bf16* p0, const __bf16* p1) {
  v8bf a = *(const v8bf*)p0;
  v8bf b = *(const v8bf*)p1;
  v16bf r;
#pragma unroll
  for (int i = 0; i < 8; ++i) { r[i] = a[i]; r[i + 8] = b[i]; }
  return r;
}

static __device__ __forceinline__ v8f wmma_bf16(v16bf a, v16bf b, v8f c) {
  return __builtin_amdgcn_wmma_f32_16x16x32_bf16(false, a, false, b, (short)0, c,
                                                 false, false);
}

#if USE_TDM
// Tensor Data Mover: 2-D tile load (element size 2B) per CDNA5 D# spec.
static __device__ __forceinline__ void tdm_load_2d(unsigned lds_addr,
                                                   const __bf16* gaddr,
                                                   unsigned tensor_d0,
                                                   unsigned tensor_d1,
                                                   unsigned tile_d0,
                                                   unsigned tile_d1,
                                                   unsigned stride0) {
  unsigned long long ga = (unsigned long long)(size_t)gaddr;
  u32x4 g0;
  g0[0] = 1u;                                   // count=1 valid descriptor
  g0[1] = lds_addr;                             // LDS byte address
  g0[2] = (unsigned)ga;                         // global_addr[31:0]
  g0[3] = (unsigned)(ga >> 32) | 0x80000000u;   // global_addr[56:32] | type=2
  i32x8 g1;
  g1[0] = 0x00010000;                           // workgroup_mask=0, data_size=1 (2B)
  g1[1] = (int)((tensor_d0 & 0xFFFFu) << 16);
  g1[2] = (int)(((tensor_d0 >> 16) & 0xFFFFu) | ((tensor_d1 & 0xFFFFu) << 16));
  g1[3] = (int)(((tensor_d1 >> 16) & 0xFFFFu) | ((tile_d0 & 0xFFFFu) << 16));
  g1[4] = (int)(tile_d1 & 0xFFFFu);             // tile_dim1, tile_dim2=0
  g1[5] = (int)stride0;                         // tensor_dim0_stride[31:0]
  g1[6] = 0;
  g1[7] = 0;
  i32x4 z4 = {};
#if __has_include(<hip/amd_detail/amd_gfx1250_TDM.h>)
  i32x8 z8 = {};
  __builtin_amdgcn_tensor_load_to_lds(g0, g1, z4, z4, z8, 0);
#else
  __builtin_amdgcn_tensor_load_to_lds(g0, g1, z4, z4, 0);
#endif
}
#endif

// ---------------- conversion kernels ----------------
__global__ void cvt_bf16_kernel(const float* __restrict__ src,
                                __bf16* __restrict__ dst, int n) {
  int i = blockIdx.x * blockDim.x + threadIdx.x;
  if (i < n) dst[i] = (__bf16)src[i];
}

// W: [H, D, E] f32  ->  Wt: [H, E, D] bf16  (so B-fragments load contiguously)
__global__ void cvt_wT_kernel(const float* __restrict__ W, __bf16* __restrict__ Wt) {
  int i = blockIdx.x * blockDim.x + threadIdx.x;  // over H*D*E (multiple of 256)
  int e = i % HD;
  int d = (i / HD) % DIM;
  int h = i / (HD * DIM);
  Wt[((size_t)h * HD + e) * DIM + d] = (__bf16)W[i];
}

// ---------------- pipelined 32x64 GEMM fragments ----------------
struct PFrag {
  v16bf a0, a1;   // two 16-row A fragments
  v16bf b[4];     // four 16-col B fragments
};

static __device__ __forceinline__ PFrag load_frags(const __bf16* arow0,
                                                   const __bf16* arow1,
                                                   const __bf16* brow, int kk0,
                                                   int half) {
  PFrag f;
  const int ca = kk0 + (half ? 8 : 0);
  f.a0 = ld2(arow0 + ca, arow0 + ca + 16);
  f.a1 = ld2(arow1 + ca, arow1 + ca + 16);
  const int rb = kk0 + (half ? 8 : 0);
#pragma unroll
  for (int nb = 0; nb < 4; ++nb) {
    const __bf16* p = brow + (size_t)nb * 16 * DIM + rb;
    f.b[nb] = ld2(p, p + 16);
  }
  return f;
}

static __device__ __forceinline__ void wmma8(const PFrag& f, v8f acc0[4],
                                             v8f acc1[4]) {
#pragma unroll
  for (int nb = 0; nb < 4; ++nb) {
    acc0[nb] = wmma_bf16(f.a0, f.b[nb], acc0[nb]);
    acc1[nb] = wmma_bf16(f.a1, f.b[nb], acc1[nb]);
  }
}

// ---------------- per-head projection GEMM ----------------
// 128-thread blocks (4 waves, 1/SIMD) + waves-per-eu=1 so the ping-pong
// fragment pipeline (~200 VGPRs) stays register-resident (no scratch spills).
__global__ void __launch_bounds__(128, 1)
proj_kernel(const __bf16* __restrict__ X,    // [NROW, DIM]
            const __bf16* __restrict__ Wt,   // [NH, HD, DIM]
            const float*  __restrict__ bias, // [NH, HD]
            __bf16* __restrict__ Out, int vTranspose) {
  const int w = threadIdx.x >> 5, lane = threadIdx.x & 31;
  const int half = lane >> 4, ln = lane & 15;
  const int m0 = (blockIdx.x * 4 + w) * 32;   // 32-row tile per wave
  const int h  = blockIdx.y;

  const __bf16* arow0 = X + (size_t)(m0 + ln) * DIM;
  const __bf16* arow1 = X + (size_t)(m0 + 16 + ln) * DIM;
  const __bf16* brow  = Wt + ((size_t)h * HD + ln) * DIM;

  v8f acc0[4] = {}, acc1[4] = {};
  // ping-pong fragment sets: each refill defines the set directly from loads,
  // so there is no loop-carried register copy.
  PFrag fa = load_frags(arow0, arow1, brow, 0, half);
  PFrag fb = load_frags(arow0, arow1, brow, 32, half);
  for (int kk0 = 0; kk0 < DIM - 64; kk0 += 64) {
    __builtin_prefetch(arow0 + kk0 + 96, 0, 1);
    wmma8(fa, acc0, acc1);
    fa = load_frags(arow0, arow1, brow, kk0 + 64, half);
    wmma8(fb, acc0, acc1);
    fb = load_frags(arow0, arow1, brow, kk0 + 96, half);
  }
  wmma8(fa, acc0, acc1);
  wmma8(fb, acc0, acc1);

#pragma unroll
  for (int nb = 0; nb < 4; ++nb) {
    const int e = nb * 16 + ln;
    const float bv = bias[h * HD + e];
#pragma unroll
    for (int half32 = 0; half32 < 2; ++half32) {
#pragma unroll
      for (int i = 0; i < 8; ++i) {
        const int gr = m0 + half32 * 16 + i + (half ? 8 : 0);
        const int b = gr >> 10, s = gr & (SEQ - 1);
        const float val = (half32 ? acc1[nb][i] : acc0[nb][i]) + bv;
        if (!vTranspose)
          Out[(((size_t)b * NH + h) * SEQ + s) * HD + e] = (__bf16)val;
        else
          Out[(((size_t)b * NH + h) * HD + e) * SEQ + s] = (__bf16)val;
      }
    }
  }
}

// ---------------- flash attention with TDM-staged K/V tiles ----------------
__global__ void attn_kernel(const __bf16* __restrict__ Qh,  // [B,H,S,HD]
                            const __bf16* __restrict__ Kh,  // [B,H,S,HD]
                            const __bf16* __restrict__ Vt,  // [B,H,HD,S]
                            __bf16* __restrict__ Ctx) {     // [NROW, DIM]
  __shared__ __align__(16) __bf16 KL[2][32 * HD];  // K tile: 32 rows x 64, 2x4KB
  __shared__ __align__(16) __bf16 VL[2][HD * 32];  // V^T tile: 64 rows x 32, 2x4KB
  __shared__ __align__(16) __bf16 Pt[8][16][32];   // per-wave P staging, 8KB
  const int tid = threadIdx.x;
  const int w = tid >> 5, lane = tid & 31;
  const int half = lane >> 4, ln = lane & 15;
  const int bh = blockIdx.x;                       // b*NH + h
  const int b = bh >> 4, h = bh & 15;
  const int m0 = (blockIdx.y * 8 + w) * 16;        // q-row tile per wave

  const __bf16* Qb = Qh + (size_t)bh * SEQ * HD;
  const __bf16* Kb = Kh + (size_t)bh * SEQ * HD;
  const __bf16* Vb = Vt + (size_t)bh * HD * SEQ;

  v16bf Qa[2];
  {
    const __bf16* qrow = Qb + (size_t)(m0 + ln) * HD;
#pragma unroll
    for (int t = 0; t < 2; ++t) {
      const int c = 32 * t + (half ? 8 : 0);
      Qa[t] = ld2(qrow + c, qrow + c + 16);
    }
  }

  v8f acc[4] = {};
  float mrow[8], lrow[8];
#pragma unroll
  for (int i = 0; i < 8; ++i) { mrow[i] = -3.0e38f; lrow[i] = 0.0f; }
  const float sm_scale = 0.125f;  // 1/sqrt(HD)

  // ---- stage first KV block into buffer 0 ----
#if USE_TDM
  if (w == 0) {
    tdm_load_2d((unsigned)(size_t)(void*)&KL[0][0], Kb, HD, SEQ, HD, 32, HD);
    tdm_load_2d((unsigned)(size_t)(void*)&VL[0][0], Vb, SEQ, HD, 32, HD, SEQ);
  }
#else
  {
    *(v8bf*)(&KL[0][0] + tid * 8) = *(const v8bf*)(Kb + tid * 8);
    const int row = tid >> 2, qd = (tid & 3) * 8;
    *(v8bf*)(&VL[0][0] + row * 32 + qd) = *(const v8bf*)(Vb + (size_t)row * SEQ + qd);
  }
#endif

  for (int kv0 = 0; kv0 < SEQ; kv0 += 32) {
    const int buf = (kv0 >> 5) & 1;
#if USE_TDM
    if (w == 0) __builtin_amdgcn_s_wait_tensorcnt(0);
#endif
    __syncthreads();  // staged buffer visible to all waves

    if (kv0 + 32 < SEQ) {  // stage next block into other buffer
#if USE_TDM
      if (w == 0) {
        tdm_load_2d((unsigned)(size_t)(void*)&KL[buf ^ 1][0],
                    Kb + (size_t)(kv0 + 32) * HD, HD, SEQ, HD, 32, HD);
        tdm_load_2d((unsigned)(size_t)(void*)&VL[buf ^ 1][0],
                    Vb + kv0 + 32, SEQ, HD, 32, HD, SEQ);
      }
#else
      {
        *(v8bf*)(&KL[buf ^ 1][0] + tid * 8) =
            *(const v8bf*)(Kb + (size_t)(kv0 + 32) * HD + tid * 8);
        const int row = tid >> 2, qd = (tid & 3) * 8;
        *(v8bf*)(&VL[buf ^ 1][0] + row * 32 + qd) =
            *(const v8bf*)(Vb + (size_t)row * SEQ + kv0 + 32 + qd);
      }
#endif
    }

    // scores: Q(16xHD) @ K^T -> two 16x16 fragments (K rows from LDS)
    v8f sc[2] = {};
#pragma unroll
    for (int nb = 0; nb < 2; ++nb) {
      const __bf16* krow = &KL[buf][(nb * 16 + ln) * HD];
#pragma unroll
      for (int t = 0; t < 2; ++t) {
        const int r = 32 * t + (half ? 8 : 0);
        v16bf kf = ld2(krow + r, krow + r + 16);
        sc[nb] = wmma_bf16(Qa[t], kf, sc[nb]);
      }
    }

    // online softmax (row stats replicated across each 16-lane half)
#pragma unroll
    for (int i = 0; i < 8; ++i) {
      float s0 = sc[0][i] * sm_scale;
      float s1 = sc[1][i] * sm_scale;
      float mx = fmaxf(s0, s1);
#pragma unroll
      for (int o = 1; o < 16; o <<= 1) mx = fmaxf(mx, __shfl_xor(mx, o, 16));
      const float mnew = fmaxf(mrow[i], mx);
      const float corr = __expf(mrow[i] - mnew);
      const float p0 = __expf(s0 - mnew);
      const float p1 = __expf(s1 - mnew);
      float rs = p0 + p1;
#pragma unroll
      for (int o = 1; o < 16; o <<= 1) rs += __shfl_xor(rs, o, 16);
      lrow[i] = lrow[i] * corr + rs;
      mrow[i] = mnew;
#pragma unroll
      for (int c = 0; c < 4; ++c) acc[c][i] *= corr;
      Pt[w][i + half * 8][ln]      = (__bf16)p0;
      Pt[w][i + half * 8][16 + ln] = (__bf16)p1;
    }
    __syncthreads();

    // P (16x32 from LDS) @ V-block (32xHD from LDS V^T tile)
    const __bf16* prow = &Pt[w][ln][0];
    const int cp = half ? 8 : 0;
    v16bf Pa = ld2(prow + cp, prow + cp + 16);
#pragma unroll
    for (int c = 0; c < 4; ++c) {
      const __bf16* vrow = &VL[buf][(c * 16 + ln) * 32];
      const int r = half ? 8 : 0;
      v16bf vf = ld2(vrow + r, vrow + r + 16);
      acc[c] = wmma_bf16(Pa, vf, acc[c]);
    }
    __syncthreads();  // all reads of buf done before it is re-staged
  }

  // normalize and scatter heads into [NROW, DIM] bf16 context
#pragma unroll
  for (int i = 0; i < 8; ++i) {
    const float inv = 1.0f / lrow[i];
    const int gr = m0 + i + (half ? 8 : 0);
#pragma unroll
    for (int c = 0; c < 4; ++c) {
      Ctx[((size_t)b * SEQ + gr) * DIM + h * HD + c * 16 + ln] =
          (__bf16)(acc[c][i] * inv);
    }
  }
}

// ---------------- output projection (32x64 tiles, ping-pong pipelined) ------
__global__ void __launch_bounds__(128, 1)
outproj_kernel(const __bf16* __restrict__ Ctx, // [NROW, DIM]
               const __bf16* __restrict__ Wo,  // [DIM, DIM]
               const float*  __restrict__ bo,  // [DIM]
               float* __restrict__ Out) {      // [NROW, DIM] f32
  const int w = threadIdx.x >> 5, lane = threadIdx.x & 31;
  const int half = lane >> 4, ln = lane & 15;
  const int tile = blockIdx.x * 4 + w;   // 2048 wave tiles
  const int m0 = (tile >> 4) * 32;       // 128 M-tiles of 32
  const int j0 = (tile & 15) * 64;       // 16 N-tiles of 64

  const __bf16* arow0 = Ctx + (size_t)(m0 + ln) * DIM;
  const __bf16* arow1 = Ctx + (size_t)(m0 + 16 + ln) * DIM;
  const __bf16* brow  = Wo + (size_t)(j0 + ln) * DIM;

  v8f acc0[4] = {}, acc1[4] = {};
  PFrag fa = load_frags(arow0, arow1, brow, 0, half);
  PFrag fb = load_frags(arow0, arow1, brow, 32, half);
  for (int kk0 = 0; kk0 < DIM - 64; kk0 += 64) {
    __builtin_prefetch(arow0 + kk0 + 96, 0, 1);
    wmma8(fa, acc0, acc1);
    fa = load_frags(arow0, arow1, brow, kk0 + 64, half);
    wmma8(fb, acc0, acc1);
    fb = load_frags(arow0, arow1, brow, kk0 + 96, half);
  }
  wmma8(fa, acc0, acc1);
  wmma8(fb, acc0, acc1);

#pragma unroll
  for (int nb = 0; nb < 4; ++nb) {
    const int j = j0 + nb * 16 + ln;
    const float bv = bo[j];
#pragma unroll
    for (int i = 0; i < 8; ++i) {
      const int r0 = m0 + i + (half ? 8 : 0);
      Out[(size_t)r0 * DIM + j] = acc0[nb][i] + bv;
      Out[(size_t)(r0 + 16) * DIM + j] = acc1[nb][i] + bv;
    }
  }
}

extern "C" void kernel_launch(void* const* d_in, const int* in_sizes, int n_in,
                              void* d_out, int out_size, void* d_ws, size_t ws_size,
                              hipStream_t stream) {
  (void)in_sizes; (void)n_in; (void)out_size; (void)ws_size;
  const float* q   = (const float*)d_in[0];
  const float* k   = (const float*)d_in[1];
  const float* v   = (const float*)d_in[2];
  const float* Wq  = (const float*)d_in[3];
  const float* bq  = (const float*)d_in[4];
  const float* Wk  = (const float*)d_in[5];
  const float* bk  = (const float*)d_in[6];
  const float* Wv  = (const float*)d_in[7];
  const float* bv  = (const float*)d_in[8];
  const float* Wo  = (const float*)d_in[9];
  const float* bo  = (const float*)d_in[10];
  float* out = (float*)d_out;

  const size_t NX = (size_t)NROW * DIM;            // 4 Mi elems
  const size_t NQ = (size_t)BATCH * NH * SEQ * HD; // 4 Mi elems
  const size_t NW = (size_t)NH * DIM * HD;         // 1 Mi elems

  __bf16* Xq  = (__bf16*)d_ws;
  __bf16* Xk  = Xq  + NX;
  __bf16* Xv  = Xk  + NX;
  __bf16* Wqt = Xv  + NX;
  __bf16* Wkt = Wqt + NW;
  __bf16* Wvt = Wkt + NW;
  __bf16* Wob = Wvt + NW;
  __bf16* Qh  = Wob + (size_t)DIM * DIM;
  __bf16* Kh  = Qh  + NQ;
  __bf16* Vt  = Kh  + NQ;                          // [B,H,HD,S]
  __bf16* Ctx = Vt  + NQ;                          // total 64 MB of d_ws

  // f32 -> bf16 staging
  cvt_bf16_kernel<<<(int)(NX / 256), 256, 0, stream>>>(q, Xq, (int)NX);
  cvt_bf16_kernel<<<(int)(NX / 256), 256, 0, stream>>>(k, Xk, (int)NX);
  cvt_bf16_kernel<<<(int)(NX / 256), 256, 0, stream>>>(v, Xv, (int)NX);
  cvt_wT_kernel<<<(int)(NW / 256), 256, 0, stream>>>(Wq, Wqt);
  cvt_wT_kernel<<<(int)(NW / 256), 256, 0, stream>>>(Wk, Wkt);
  cvt_wT_kernel<<<(int)(NW / 256), 256, 0, stream>>>(Wv, Wvt);
  cvt_bf16_kernel<<<DIM * DIM / 256, 256, 0, stream>>>(Wo, Wob, DIM * DIM);

  // Q/K/V projections (V stored transposed for contiguous P@V fragments)
  dim3 gp(NROW / 128, NH);   // 32 rows/wave * 4 waves/block
  proj_kernel<<<gp, 128, 0, stream>>>(Xq, Wqt, bq, Qh, 0);
  proj_kernel<<<gp, 128, 0, stream>>>(Xk, Wkt, bk, Kh, 0);
  proj_kernel<<<gp, 128, 0, stream>>>(Xv, Wvt, bv, Vt, 1);

  // flash attention (TDM-staged KV tiles when available)
  attn_kernel<<<dim3(BATCH * NH, SEQ / 128), 256, 0, stream>>>(Qh, Kh, Vt, Ctx);

  // final projection to f32 output
  outproj_kernel<<<(NROW / 32) * (DIM / 64) / 4, 128, 0, stream>>>(Ctx, Wob, bo, out);
}
